// LocalAttention_51049981281199
// MI455X (gfx1250) — matrix operand order repaired and checked by reading, not compiled, not run on hardware
//
#include <hip/hip_runtime.h>

// ---------------- problem constants ----------------
#define BB 2
#define SS 2048
#define DD 1024
#define HH 16
#define HD 64
#define SPAD (SS + 192)   // 64 left pad + 128 right pad for the sliding V window
#define MM (BB * SS)      // 4096 GEMM rows

typedef __attribute__((ext_vector_type(16))) _Float16 v16h;
typedef __attribute__((ext_vector_type(8)))  _Float16 v8h;
typedef __attribute__((ext_vector_type(4)))  _Float16 v4h;
typedef __attribute__((ext_vector_type(8)))  float    v8f;

__device__ __forceinline__ v16h pack16(v8h lo, v8h hi) {
  v16h r;
#pragma unroll
  for (int i = 0; i < 8; ++i) { r[i] = lo[i]; r[i + 8] = hi[i]; }
  return r;
}

// A-fragment (16x32 f16): caller passes this lane's row pointer (row = lane%16).
// halves 0-15 = K[kb + grp*16 .. +15] contiguous.
__device__ __forceinline__ v16h load_a_frag(const _Float16* rowp, int kb, int grp) {
  v8h lo = *(const v8h*)(rowp + kb + grp * 16);
  v8h hi = *(const v8h*)(rowp + kb + grp * 16 + 8);
  return pack16(lo, hi);
}

// B-fragment (32x16 f16): caller passes the K-source row for this lane's column
// (col = lane%16). halves 0-15 = K[kb + grp*16 .. +15] contiguous.
__device__ __forceinline__ v16h load_b_frag(const _Float16* rowp, int kb, int grp) {
  v8h lo = *(const v8h*)(rowp + kb + grp * 16);
  v8h hi = *(const v8h*)(rowp + kb + grp * 16 + 8);
  return pack16(lo, hi);
}

__device__ __forceinline__ v8f wmma_f16(v16h a, v16h b, v8f c) {
  return __builtin_amdgcn_wmma_f32_16x16x32_f16(false, a, false, b, (short)0, c,
                                                false, false);
}

// ---------------- helper kernels ----------------
__global__ void __launch_bounds__(256) f32_to_f16_kernel(
    const float* __restrict__ src, _Float16* __restrict__ dst, int n4) {
  int i = blockIdx.x * blockDim.x + threadIdx.x;
  int stride = gridDim.x * blockDim.x;
  for (; i < n4; i += stride) {
    float4 v = ((const float4*)src)[i];
    v4h h = {( _Float16)v.x, (_Float16)v.y, (_Float16)v.z, (_Float16)v.w};
    ((v4h*)dst)[i] = h;
  }
}

__global__ void __launch_bounds__(256) zero16_kernel(float4* __restrict__ p, int n16) {
  int i = blockIdx.x * blockDim.x + threadIdx.x;
  int stride = gridDim.x * blockDim.x;
  float4 z = make_float4(0.f, 0.f, 0.f, 0.f);
  for (; i < n16; i += stride) p[i] = z;
}

// ---------------- GEMM: out[m,n] = A[m,:] . Bw[n,:] + bias[n] ----------------
// A: [MM, DD] f16 row-major, Bw: [DD, DD] f16 row-major (Linear weight, NT gemm).
// mode 0: f16 store to dst[m*DD + n]         (Q, K)
// mode 1: f16 store transposed-per-head Vt[( (b*HH+h)*HD + d )*SPAD + 64 + s]
// mode 2: f32 store to dst[m*DD + n]         (final out)
__global__ void __launch_bounds__(32) gemm64_kernel(
    const _Float16* __restrict__ A, const _Float16* __restrict__ Bw,
    const float* __restrict__ bias, void* __restrict__ dst, int mode) {
  const int lane = threadIdx.x;
  const int idx = lane & 15, grp = lane >> 4;
  const int m0 = blockIdx.x * 64;
  const int n0 = blockIdx.y * 64;

  v8f acc[4][4];
#pragma unroll
  for (int im = 0; im < 4; ++im)
#pragma unroll
    for (int in = 0; in < 4; ++in) acc[im][in] = (v8f)(0.f);

  const _Float16* arow[4];
  const _Float16* brow[4];
#pragma unroll
  for (int im = 0; im < 4; ++im) arow[im] = A + (size_t)(m0 + im * 16 + idx) * DD;
#pragma unroll
  for (int in = 0; in < 4; ++in) brow[in] = Bw + (size_t)(n0 + in * 16 + idx) * DD;

  for (int k = 0; k < DD; k += 32) {
    v16h af[4], bf[4];
#pragma unroll
    for (int im = 0; im < 4; ++im) af[im] = load_a_frag(arow[im], k, grp);
#pragma unroll
    for (int in = 0; in < 4; ++in) bf[in] = load_b_frag(brow[in], k, grp);
#pragma unroll
    for (int im = 0; im < 4; ++im)
#pragma unroll
      for (int in = 0; in < 4; ++in)
        acc[im][in] = wmma_f16(af[im], bf[in], acc[im][in]);
  }

#pragma unroll
  for (int in = 0; in < 4; ++in) {
    float bn = bias[n0 + in * 16 + idx];
#pragma unroll
    for (int im = 0; im < 4; ++im)
#pragma unroll
      for (int r = 0; r < 8; ++r) acc[im][in][r] += bn;
  }

  if (mode == 0) {
    _Float16* o = (_Float16*)dst;
#pragma unroll
    for (int im = 0; im < 4; ++im)
#pragma unroll
      for (int in = 0; in < 4; ++in) {
        const int n = n0 + in * 16 + idx;
        const int mrow = m0 + im * 16 + grp * 8;
        _Float16* p = o + (size_t)mrow * DD + n;
#pragma unroll
        for (int r = 0; r < 8; ++r) p[(size_t)r * DD] = (_Float16)acc[im][in][r];
      }
  } else if (mode == 1) {
    _Float16* o = (_Float16*)dst;
#pragma unroll
    for (int im = 0; im < 4; ++im)
#pragma unroll
      for (int in = 0; in < 4; ++in) {
        const int e = n0 + in * 16 + idx;
        const int h = e >> 6, d = e & 63;
        const int mrow = m0 + im * 16 + grp * 8;
        const int b = mrow >> 11, s = mrow & (SS - 1);
        v8h h8;
#pragma unroll
        for (int r = 0; r < 8; ++r) h8[r] = (_Float16)acc[im][in][r];
        *(v8h*)(o + (size_t)((b * HH + h) * HD + d) * SPAD + 64 + s) = h8;
      }
  } else {
    float* o = (float*)dst;
#pragma unroll
    for (int im = 0; im < 4; ++im)
#pragma unroll
      for (int in = 0; in < 4; ++in) {
        const int n = n0 + in * 16 + idx;
        const int mrow = m0 + im * 16 + grp * 8;
        float* p = o + (size_t)mrow * DD + n;
#pragma unroll
        for (int r = 0; r < 8; ++r) p[(size_t)r * DD] = acc[im][in][r];
      }
  }
}

// ---------------- banded attention: 1 wave per (b, h, 16-row q tile) ----------
__global__ void __launch_bounds__(32) local_attn_kernel(
    const _Float16* __restrict__ Qh, const _Float16* __restrict__ Kh,
    const _Float16* __restrict__ Vt, float* __restrict__ attn,
    _Float16* __restrict__ ctxh) {
  __shared__ _Float16 plds[16][160];

  const int lane = threadIdx.x;
  const int idx = lane & 15, grp = lane >> 4;
  int blk = blockIdx.x;
  const int qt = blk & (SS / 16 - 1);
  blk >>= 7;
  const int h = blk & (HH - 1);
  const int b = blk >> 4;
  const int s0 = qt * 16;

  // Q A-fragments for this tile (row = s0+idx, d in [0,64))
  const _Float16* qrow = Qh + (size_t)(b * SS + s0 + idx) * DD + h * HD;
  const v16h a0 = load_a_frag(qrow, 0, grp);
  const v16h a1 = load_a_frag(qrow, 32, grp);

  // scores over 9 column tiles covering [s0-64, s0+80)
  float sc[9][8];
#pragma unroll
  for (int t = 0; t < 9; ++t) {
    const int j0 = s0 - 64 + 16 * t;
    int jc = j0 + idx;
    jc = jc < 0 ? 0 : (jc > SS - 1 ? SS - 1 : jc);
    const _Float16* krow = Kh + (size_t)(b * SS + jc) * DD + h * HD;
    v16h b0 = load_b_frag(krow, 0, grp);
    v16h b1 = load_b_frag(krow, 32, grp);
    v8f s = (v8f)(0.f);
    s = wmma_f16(a0, b0, s);
    s = wmma_f16(a1, b1, s);
#pragma unroll
    for (int r = 0; r < 8; ++r) sc[t][r] = s[r];
  }

  // mask + scale: element (i = s0+r+8*grp, j = s0-64+16t+idx)
#pragma unroll
  for (int t = 0; t < 9; ++t)
#pragma unroll
    for (int r = 0; r < 8; ++r) {
      const int irel = r + 8 * grp;
      const int jrel = -64 + 16 * t + idx;
      const int j = s0 + jrel;
      const int dlt = irel - jrel;
      const bool ok = (j >= 0) & (j < SS) & (dlt <= 64) & (dlt >= -64);
      sc[t][r] = ok ? sc[t][r] * 0.125f : -1.0e30f;
    }

  // softmax per row (144 values spread: 16 lanes of this group x 9 tiles)
#pragma unroll
  for (int r = 0; r < 8; ++r) {
    float m = sc[0][r];
#pragma unroll
    for (int t = 1; t < 9; ++t) m = fmaxf(m, sc[t][r]);
#pragma unroll
    for (int off = 1; off < 16; off <<= 1) m = fmaxf(m, __shfl_xor(m, off, 32));
    float sum = 0.f;
#pragma unroll
    for (int t = 0; t < 9; ++t) {
      float e = __expf(sc[t][r] - m);
      sc[t][r] = e;
      sum += e;
    }
#pragma unroll
    for (int off = 1; off < 16; off <<= 1) sum += __shfl_xor(sum, off, 32);
    const float inv = 1.0f / sum;
#pragma unroll
    for (int t = 0; t < 9; ++t) sc[t][r] *= inv;
  }

  // scatter band probabilities to the dense attn output + stage f16 in LDS
  float* abase = attn + ((size_t)(b * HH + h) * SS + s0) * SS;
#pragma unroll
  for (int t = 0; t < 9; ++t)
#pragma unroll
    for (int r = 0; r < 8; ++r) {
      const int irel = r + 8 * grp;
      const int j = s0 - 64 + 16 * t + idx;
      if (j >= 0 && j < SS) abase[(size_t)irel * SS + j] = sc[t][r];
      plds[irel][16 * t + idx] = (_Float16)sc[t][r];
    }
  // zero pad columns 144..159 (K padding for the 5th ctx chunk)
#pragma unroll
  for (int z = 0; z < 8; ++z) {
    int q = z * 32 + lane;  // 256 slots
    plds[q >> 4][144 + (q & 15)] = (_Float16)0.f;
  }
  __syncthreads();

  // ctx = P (16x160, zero-padded) @ Vwindow (160x64), Vt padded so always in-bounds
  v8f c[4];
#pragma unroll
  for (int n = 0; n < 4; ++n) c[n] = (v8f)(0.f);
  const _Float16* vhead = Vt + (size_t)(b * HH + h) * HD * SPAD;
#pragma unroll
  for (int kc = 0; kc < 5; ++kc) {
    v16h ap = load_a_frag(&plds[idx][0], 32 * kc, grp);
#pragma unroll
    for (int n = 0; n < 4; ++n) {
      const int d = n * 16 + idx;
      // padded col = 64 + (s0-64) + 32*kc + grp*16 = s0 + 32*kc + grp*16
      const _Float16* vrow = vhead + (size_t)d * SPAD + s0 + 32 * kc + grp * 16;
      v8h lo = *(const v8h*)vrow;
      v8h hi = *(const v8h*)(vrow + 8);
      c[n] = wmma_f16(ap, pack16(lo, hi), c[n]);
    }
  }

  // store ctx f16 [B*S, D] for the output projection
#pragma unroll
  for (int n = 0; n < 4; ++n)
#pragma unroll
    for (int r = 0; r < 8; ++r) {
      const int irel = r + 8 * grp;
      ctxh[(size_t)(b * SS + s0 + irel) * DD + h * HD + n * 16 + idx] =
          (_Float16)c[n][r];
    }
}

// ---------------- host launcher ----------------
extern "C" void kernel_launch(void* const* d_in, const int* in_sizes, int n_in,
                              void* d_out, int out_size, void* d_ws, size_t ws_size,
                              hipStream_t stream) {
  (void)in_sizes; (void)n_in; (void)out_size; (void)ws_size;
  const float* x  = (const float*)d_in[0];
  const float* Wq = (const float*)d_in[1];
  const float* bq = (const float*)d_in[2];
  const float* Wk = (const float*)d_in[3];
  const float* bk = (const float*)d_in[4];
  const float* Wv = (const float*)d_in[5];
  const float* bv = (const float*)d_in[6];
  const float* Wo = (const float*)d_in[7];
  const float* bo = (const float*)d_in[8];

  float* out  = (float*)d_out;                       // [B,S,D]
  float* attn = out + (size_t)BB * SS * DD;          // [B,H,S,S]

  constexpr size_t SZ_X = (size_t)MM * DD * 2;       // f16 x / Q / K / ctx
  constexpr size_t SZ_W = (size_t)DD * DD * 2;       // f16 weight
  constexpr size_t SZ_V = (size_t)BB * HH * HD * SPAD * 2;
  char* ws = (char*)d_ws;
  _Float16* xh  = (_Float16*)(ws);
  _Float16* wqh = (_Float16*)(ws + SZ_X);
  _Float16* wkh = (_Float16*)(ws + SZ_X + SZ_W);
  _Float16* wvh = (_Float16*)(ws + SZ_X + 2 * SZ_W);
  _Float16* woh = (_Float16*)(ws + SZ_X + 3 * SZ_W);
  _Float16* Qh  = (_Float16*)(ws + SZ_X + 4 * SZ_W);
  _Float16* Kh  = (_Float16*)(ws + 2 * SZ_X + 4 * SZ_W);
  _Float16* Vt  = (_Float16*)(ws + 3 * SZ_X + 4 * SZ_W);
  _Float16* Ch  = (_Float16*)(ws + 3 * SZ_X + 4 * SZ_W + SZ_V);

  // fp32 -> f16 staging
  f32_to_f16_kernel<<<dim3(1024), dim3(256), 0, stream>>>(x,  xh,  MM * DD / 4);
  f32_to_f16_kernel<<<dim3(256),  dim3(256), 0, stream>>>(Wq, wqh, DD * DD / 4);
  f32_to_f16_kernel<<<dim3(256),  dim3(256), 0, stream>>>(Wk, wkh, DD * DD / 4);
  f32_to_f16_kernel<<<dim3(256),  dim3(256), 0, stream>>>(Wv, wvh, DD * DD / 4);
  f32_to_f16_kernel<<<dim3(256),  dim3(256), 0, stream>>>(Wo, woh, DD * DD / 4);

  // zero the dense attn output (band gets overwritten) and padded Vt
  zero16_kernel<<<dim3(4096), dim3(256), 0, stream>>>((float4*)attn,
      (int)((size_t)BB * HH * SS * SS * 4 / 16));
  zero16_kernel<<<dim3(512), dim3(256), 0, stream>>>((float4*)Vt, (int)(SZ_V / 16));

  // projections (WMMA f16, fp32 accumulate)
  dim3 gg(MM / 64, DD / 64);
  gemm64_kernel<<<gg, 32, 0, stream>>>(xh, wqh, bq, Qh, 0);
  gemm64_kernel<<<gg, 32, 0, stream>>>(xh, wkh, bk, Kh, 0);
  gemm64_kernel<<<gg, 32, 0, stream>>>(xh, wvh, bv, Vt, 1);

  // banded attention
  local_attn_kernel<<<dim3(BB * HH * (SS / 16)), 32, 0, stream>>>(Qh, Kh, Vt, attn, Ch);

  // output projection -> fp32
  gemm64_kernel<<<gg, 32, 0, stream>>>(Ch, woh, bo, out, 2);
}
